// DigitConvolutionalModel_25864293056599
// MI455X (gfx1250) — matrix operand order repaired and checked by reading, not compiled, not run on hardware
//
#include <hip/hip_runtime.h>

typedef __attribute__((ext_vector_type(16))) _Float16 v16h;
typedef __attribute__((ext_vector_type(8)))  _Float16 v8h;
typedef __attribute__((ext_vector_type(8)))  float    v8f;

constexpr int BATCH   = 65536;
constexpr int M_TILE  = 64;    // batch rows per block (4 strips of 16)
constexpr int THREADS = 128;   // 4 wave32s
constexpr int K1      = 676;   // 26*26 conv features
constexpr int K1P     = 704;   // padded K: 22 WMMA K-steps
constexpr int K1LD    = 712;   // sFeat row stride (halves): 16B aligned, bank-conflict free
constexpr int N1      = 300;
constexpr int N1P     = 320;   // 20 N-tiles; also GEMM2 K (10 steps)
constexpr int N1LD    = 328;   // sH1 row stride (halves): 16B aligned, bank-conflict free
constexpr int N2      = 10;
constexpr int KT1     = K1P / 32;   // 22
constexpr int NT1     = N1P / 16;   // 20
constexpr int KT2     = N1P / 32;   // 10
constexpr int FRAG    = 512;        // halves per 32x16 B fragment (32 lanes * 16)
constexpr int W1PACK_ELEMS = NT1 * KT1 * FRAG;   // 225,280 halves = 450,560 B

// ---- Prologue: pack W1 (fp32, [676][300]) into fragment-major f16 ----
// dst[(nt*KT1 + kt)*512 + lane*16 + j] = w1[kt*32 + (lane>>4)*16 + j][nt*16 + (lane&15)]
// so each lane's B fragment is one contiguous 32-byte run.
__global__ __launch_bounds__(256)
void pack_w1_kernel(const float* __restrict__ w1, _Float16* __restrict__ dst)
{
    int idx  = blockIdx.x * 256 + threadIdx.x;
    if (idx >= W1PACK_ELEMS) return;
    int j    = idx & 15;
    int l    = (idx >> 4) & 31;
    int frag = idx >> 9;
    int nt   = frag / KT1;
    int kt   = frag - nt * KT1;
    int k    = kt * 32 + (l >> 4) * 16 + j;
    int n    = nt * 16 + (l & 15);
    float v  = 0.f;
    if (k < K1 && n < N1) v = w1[k * N1 + n];
    dst[idx] = (_Float16)v;
}

__global__ __launch_bounds__(THREADS)
void fused_conv_mlp_wmma(const float* __restrict__ x,
                         const float* __restrict__ convw,
                         const _Float16* __restrict__ w1pack,
                         const float* __restrict__ b1,
                         const float* __restrict__ w2,
                         const float* __restrict__ b2,
                         float* __restrict__ out)
{
    extern __shared__ char smemRaw[];
    _Float16* sFeat = (_Float16*)smemRaw;            // [M_TILE][K1LD]  f16 conv features
    _Float16* sH1   = sFeat + M_TILE * K1LD;         // [M_TILE][N1LD]  f16 hidden acts
    _Float16* sW2f  = sH1   + M_TILE * N1LD;         // [KT2][512]      W2 fragments
    float*    sB1   = (float*)(sW2f + KT2 * FRAG);   // [N1P]
    float*    sB2   = sB1 + N1P;                     // [16]

    const int t      = threadIdx.x;
    const int blkRow = blockIdx.x * M_TILE;

    float cw[9];
    #pragma unroll
    for (int i = 0; i < 9; ++i) cw[i] = convw[i];

    // ---- Phase A: 3x3 valid conv (fp32) -> f16 feature tile in LDS ----
    const float* xblk = x + (size_t)blkRow * 784;
    for (int idx = t; idx < M_TILE * K1P; idx += THREADS) {
        int m = idx / K1P;
        int k = idx - m * K1P;
        _Float16 v = (_Float16)0.f;
        if (k < K1) {
            int i = k / 26;
            int j = k - i * 26;
            const float* img = xblk + m * 784 + i * 28 + j;
            __builtin_prefetch(img + 784, 0, 0);   // -> global_prefetch_b8
            float s = 0.f;
            #pragma unroll
            for (int di = 0; di < 3; ++di)
                #pragma unroll
                for (int dj = 0; dj < 3; ++dj)
                    s = fmaf(img[di * 28 + dj], cw[di * 3 + dj], s);
            v = (_Float16)s;
        }
        sFeat[m * K1LD + k] = v;
    }
    // Stage W2 in fragment order (f16, zero-padded to 320x16) + biases
    for (int idx = t; idx < KT2 * FRAG; idx += THREADS) {
        int j  = idx & 15;
        int l  = (idx >> 4) & 31;
        int kt = idx >> 9;
        int k  = kt * 32 + ((l >> 4) & 1) * 16 + j;
        int n  = l & 15;
        sW2f[idx] = (k < N1 && n < N2) ? (_Float16)w2[k * N2 + n] : (_Float16)0.f;
    }
    for (int idx = t; idx < N1P; idx += THREADS)
        sB1[idx] = (idx < N1) ? b1[idx] : 0.f;
    if (t < 16) sB2[t] = (t < N2) ? b2[t] : 0.f;
    __syncthreads();

    const int lane = t & 31;
    const int wv   = t >> 5;        // wave id 0..3
    const int nl   = lane & 15;     // N col (B/C/D) and M row (A) sub-index
    const int hi   = lane >> 4;     // half-wave select

    // ---- Phase B: h1 = relu(feat @ W1 + b1) ----
    // Wave wv handles N-tiles [5*wv, 5*wv+5) for ALL 4 row strips:
    // each B fragment (1 global b128x2) feeds 4 WMMAs.
    for (int nti = 0; nti < NT1 / 4; ++nti) {
        const int nt = wv * (NT1 / 4) + nti;
        const int n  = nt * 16 + nl;
        const float bias = sB1[n];
        v8f acc[4];
        #pragma unroll
        for (int s = 0; s < 4; ++s)
            #pragma unroll
            for (int r = 0; r < 8; ++r) acc[s][r] = bias;

        const _Float16* fp = w1pack + (size_t)(nt * KT1) * FRAG + lane * 16;
        for (int kt = 0; kt < KT1; ++kt) {
            v16h bf = *(const v16h*)(fp + kt * FRAG);
            const int k0 = kt * 32;
            #pragma unroll
            for (int s = 0; s < 4; ++s) {
                const _Float16* ap = sFeat + (s * 16 + nl) * K1LD + k0 + hi * 16;
                v8h alo = *(const v8h*)(ap);
                v8h ahi = *(const v8h*)(ap + 8);
                v16h a  = __builtin_shufflevector(alo, ahi,
                              0,1,2,3,4,5,6,7,8,9,10,11,12,13,14,15);
                acc[s] = __builtin_amdgcn_wmma_f32_16x16x32_f16(false, a, false, bf,
                                                                (short)0, acc[s], false, false);
            }
        }
        #pragma unroll
        for (int s = 0; s < 4; ++s)
            #pragma unroll
            for (int r = 0; r < 8; ++r) {
                int m = s * 16 + r + hi * 8;
                float h = acc[s][r] > 0.f ? acc[s][r] : 0.f;
                sH1[m * N1LD + n] = (n < N1) ? (_Float16)h : (_Float16)0.f;
            }
    }
    __syncthreads();

    // ---- Phase C: out = h1 @ W2 + b2 (wave wv owns row strip wv) ----
    {
        const float bias = sB2[nl];
        v8f acc;
        #pragma unroll
        for (int r = 0; r < 8; ++r) acc[r] = bias;

        for (int kt = 0; kt < KT2; ++kt) {
            v16h bf = *(const v16h*)(sW2f + kt * FRAG + lane * 16);
            const int k0 = kt * 32;
            const _Float16* ap = sH1 + (wv * 16 + nl) * N1LD + k0 + hi * 16;
            v8h alo = *(const v8h*)(ap);
            v8h ahi = *(const v8h*)(ap + 8);
            v16h a  = __builtin_shufflevector(alo, ahi,
                          0,1,2,3,4,5,6,7,8,9,10,11,12,13,14,15);
            acc = __builtin_amdgcn_wmma_f32_16x16x32_f16(false, a, false, bf,
                                                         (short)0, acc, false, false);
        }
        if (nl < N2) {
            #pragma unroll
            for (int r = 0; r < 8; ++r) {
                int m = blkRow + wv * 16 + r + hi * 8;
                out[(size_t)m * N2 + nl] = acc[r];
            }
        }
    }
}

extern "C" void kernel_launch(void* const* d_in, const int* in_sizes, int n_in,
                              void* d_out, int out_size, void* d_ws, size_t ws_size,
                              hipStream_t stream) {
    (void)in_sizes; (void)n_in; (void)out_size; (void)ws_size;
    const float* x  = (const float*)d_in[0];
    const float* cw = (const float*)d_in[1];
    const float* w1 = (const float*)d_in[2];
    const float* b1 = (const float*)d_in[3];
    const float* w2 = (const float*)d_in[4];
    const float* b2 = (const float*)d_in[5];
    float* out = (float*)d_out;
    _Float16* w1pack = (_Float16*)d_ws;   // needs 450,560 B of workspace

    // Prologue: fragment-pack W1 (same stream -> ordered before the fused kernel)
    hipLaunchKernelGGL(pack_w1_kernel,
                       dim3((W1PACK_ELEMS + 255) / 256), dim3(256), 0, stream,
                       w1, w1pack);

    size_t shmem = (size_t)(M_TILE * K1LD + M_TILE * N1LD + KT2 * FRAG) * sizeof(_Float16)
                 + (size_t)(N1P + 16) * sizeof(float);   // ~141 KB

    hipLaunchKernelGGL(fused_conv_mlp_wmma,
                       dim3(BATCH / M_TILE), dim3(THREADS), shmem, stream,
                       x, cw, w1pack, b1, w2, b2, out);
}
